// NystromAttention_4587025072208
// MI455X (gfx1250) — matrix operand order repaired and checked by reading, not compiled, not run on hardware
//
#include <hip/hip_runtime.h>

typedef __bf16 bf16_t;
typedef __attribute__((ext_vector_type(8)))  __bf16 v8bf;
typedef __attribute__((ext_vector_type(16))) __bf16 v16bf;
typedef __attribute__((ext_vector_type(8)))  float  v8f;

#define B_   4
#define N_   4096
#define DIM_ 512
#define H_   8
#define D_   64
#define M_   256
#define L_   16
#define BH_  (B_*H_)
#define NT_  (B_*N_)          // 16384 total rows
#define INNER_ (H_*D_)        // 512
#define SCALE_ 0.125f         // 64^-0.5

#define WMMA_BF16(a, b, c) \
  __builtin_amdgcn_wmma_f32_16x16x32_bf16(false, (a), false, (b), (short)0, (c), false, false)

// A fragment: two 8-half runs at +0 and +16 (K = kb..kb+7 and 16+kb..16+kb+7)
__device__ inline v16bf ld_fragA(const bf16_t* p) {
  v8bf x0 = *(const v8bf*)p;
  v8bf x1 = *(const v8bf*)(p + 16);
  return __builtin_shufflevector(x0, x1, 0,1,2,3,4,5,6,7,8,9,10,11,12,13,14,15);
}
// B fragment: two 8-half runs at +0 and +16 for one column of B^T (same K order as A)
__device__ inline v16bf ld_fragB(const bf16_t* p) {
  v8bf x0 = *(const v8bf*)p;
  v8bf x1 = *(const v8bf*)(p + 16);
  return __builtin_shufflevector(x0, x1, 0,1,2,3,4,5,6,7,8,9,10,11,12,13,14,15);
}

// ---------------- WMMA core: C(32x64,f32) = [A0;A1](32xK) * [Bt0..Bt3](16xK each)^T
// 2x4 register blocking + software-pipelined (double-buffered) K loop.
__device__ inline void wmma_core_2x4(const bf16_t* __restrict__ A, int lda,
                                     const bf16_t* __restrict__ Bt, int ldb, int K,
                                     v8f acc[2][4]) {
  const int lane = threadIdx.x & 31;
  const int hi   = lane >> 4;        // 0 / 1
  const int mr   = lane & 15;
  const bf16_t* ar0 = A + (long)mr * lda + hi * 8;
  const bf16_t* ar1 = ar0 + (long)16 * lda;
  const bf16_t* br  = Bt + (long)mr * ldb + hi * 8;
  const long bs = (long)16 * ldb;
  // prologue: fragments for k = 0
  v16bf a0 = ld_fragA(ar0), a1 = ld_fragA(ar1);
  v16bf b0 = ld_fragB(br),          b1 = ld_fragB(br + bs);
  v16bf b2 = ld_fragB(br + 2 * bs), b3 = ld_fragB(br + 3 * bs);
  for (int k = 0; k < K; k += 32) {
    const int kn = (k + 32 < K) ? (k + 32) : k;   // tail: redundant reload, no OOB
    // issue next iteration's loads before consuming current fragments
    v16bf na0 = ld_fragA(ar0 + kn), na1 = ld_fragA(ar1 + kn);
    v16bf nb0 = ld_fragB(br + kn),          nb1 = ld_fragB(br + bs + kn);
    v16bf nb2 = ld_fragB(br + 2 * bs + kn), nb3 = ld_fragB(br + 3 * bs + kn);
    __builtin_prefetch(ar0 + k + 1024, 0, 3);
    acc[0][0] = WMMA_BF16(a0, b0, acc[0][0]);
    acc[0][1] = WMMA_BF16(a0, b1, acc[0][1]);
    acc[0][2] = WMMA_BF16(a0, b2, acc[0][2]);
    acc[0][3] = WMMA_BF16(a0, b3, acc[0][3]);
    acc[1][0] = WMMA_BF16(a1, b0, acc[1][0]);
    acc[1][1] = WMMA_BF16(a1, b1, acc[1][1]);
    acc[1][2] = WMMA_BF16(a1, b2, acc[1][2]);
    acc[1][3] = WMMA_BF16(a1, b3, acc[1][3]);
    a0 = na0; a1 = na1; b0 = nb0; b1 = nb1; b2 = nb2; b3 = nb3;
  }
}

__device__ inline float wave_max(float v){ for(int o=16;o;o>>=1) v=fmaxf(v,__shfl_xor(v,o,32)); return v; }
__device__ inline float wave_sum(float v){ for(int o=16;o;o>>=1) v+=__shfl_xor(v,o,32); return v; }

// ---------------- elementwise prep
__global__ void k_cvt_x(const float* __restrict__ x, bf16_t* __restrict__ xb) {
  long i = (long)blockIdx.x * 256 + threadIdx.x;                 // 16384*512
  xb[i] = (bf16_t)x[i];
}
__global__ void k_tr_w(const float* __restrict__ w, bf16_t* __restrict__ wt) {
  long i = (long)blockIdx.x * 256 + threadIdx.x;                 // out [1536][512]
  int kk = i & 511; long j = i >> 9;
  wt[i] = (bf16_t)w[(long)kk * 1536 + j];
}
__global__ void k_tr_wout(const float* __restrict__ w, bf16_t* __restrict__ wt) {
  long i = (long)blockIdx.x * 256 + threadIdx.x;                 // out [512][512]
  int kk = i & 511; long j = i >> 9;
  wt[i] = (bf16_t)w[(long)kk * 512 + j];
}

// ---------------- QKV projection: [16384x512] x [512x1536] -> q(scaled)/k/v in [bh][n][d], + v^T
__global__ __launch_bounds__(256, 1)
void k_gemm_qkv(const bf16_t* __restrict__ xb, const bf16_t* __restrict__ wqT,
                bf16_t* __restrict__ qb, bf16_t* __restrict__ kbuf,
                bf16_t* __restrict__ vbuf, bf16_t* __restrict__ vT) {
  long t = (long)blockIdx.x * 8 + (threadIdx.x >> 5);            // 512*24 wide tiles
  int tn4 = (int)(t % 24), tm2 = (int)(t / 24);
  const bf16_t* A  = xb  + (long)tm2 * 32 * 512;
  const bf16_t* Bt = wqT + (long)tn4 * 64 * 512;
  v8f acc[2][4] = {};
  wmma_core_2x4(A, 512, Bt, 512, 512, acc);
  int lane = threadIdx.x & 31, hi2 = lane >> 4, mr = lane & 15;
#pragma unroll
  for (int u = 0; u < 2; ++u) {
#pragma unroll
    for (int s = 0; s < 4; ++s) {
      int j = (tn4 * 4 + s) * 16 + mr;
      int which = j >> 9, hd = j & 511, head = hd >> 6, dc = hd & 63;
#pragma unroll
      for (int r = 0; r < 8; ++r) {
        int i  = tm2 * 32 + u * 16 + hi2 * 8 + r;
        int bi = i >> 12, ni = i & 4095;
        long off = (((long)(bi * 8 + head)) * 4096 + ni) * 64 + dc;
        float v = acc[u][s][r];
        if (which == 0)      qb[off]   = (bf16_t)(v * SCALE_);
        else if (which == 1) kbuf[off] = (bf16_t)v;
        else {
          bf16_t bv = (bf16_t)v;
          vbuf[off] = bv;
          vT[(((long)(bi * 8 + head)) * 64 + dc) * 4096 + ni] = bv;
        }
      }
    }
  }
}

// ---------------- landmark means over chunks of 16
__global__ void k_landmarks(const bf16_t* __restrict__ qb, const bf16_t* __restrict__ kb,
                            bf16_t* __restrict__ ql, bf16_t* __restrict__ kl) {
  long idx = (long)blockIdx.x * 256 + threadIdx.x;               // 32*256*64
  int dc = idx & 63; int j = (int)((idx >> 6) & 255); int bh = (int)(idx >> 14);
  long base = (((long)bh * 4096) + j * 16) * 64 + dc;
  float sq = 0.f, sk = 0.f;
#pragma unroll
  for (int tt = 0; tt < 16; ++tt) { sq += (float)qb[base + tt * 64]; sk += (float)kb[base + tt * 64]; }
  ql[idx] = (bf16_t)(sq * (1.f / 16.f));
  kl[idx] = (bf16_t)(sk * (1.f / 16.f));
}

// ---------------- generic batched GEMM -> f32 row-major (Mt2 = M tiles/2, Nt4 = N tiles/4)
__global__ __launch_bounds__(256, 1)
void k_gemm_f32(const bf16_t* __restrict__ Aall, const bf16_t* __restrict__ Ball,
                float* __restrict__ Call, int Mt2, int Nt4, int K,
                int lda, int ldb, int ldc, long sA, long sB, long sC) {
  long t = (long)blockIdx.x * 8 + (threadIdx.x >> 5);
  int per = Mt2 * Nt4;
  int b = (int)(t / per); int rem = (int)(t % per);
  int tm2 = rem / Nt4, tn4 = rem % Nt4;
  const bf16_t* A  = Aall + b * sA + (long)tm2 * 32 * lda;
  const bf16_t* Bt = Ball + b * sB + (long)tn4 * 64 * ldb;
  v8f acc[2][4] = {};
  wmma_core_2x4(A, lda, Bt, ldb, K, acc);
  int lane = threadIdx.x & 31, hi2 = lane >> 4, mr = lane & 15;
  float* C = Call + b * sC;
#pragma unroll
  for (int u = 0; u < 2; ++u) {
#pragma unroll
    for (int s = 0; s < 4; ++s) {
      int col = (tn4 * 4 + s) * 16 + mr;
#pragma unroll
      for (int r = 0; r < 8; ++r)
        C[(long)(tm2 * 32 + u * 16 + hi2 * 8 + r) * ldc + col] = acc[u][s][r];
    }
  }
}

// ---------------- generic batched GEMM -> bf16 row-major and/or transposed, (s*acc + d*I)
__global__ __launch_bounds__(256, 1)
void k_gemm_b16(const bf16_t* __restrict__ Aall, const bf16_t* __restrict__ Ball,
                bf16_t* __restrict__ RM, bf16_t* __restrict__ TR,
                int Mt2, int Nt4, int K, int lda, int ldb,
                long sA, long sB, long sRM, int ldrm, long sTR, int ldtr,
                float s_rm, float d_rm, float s_tr, float d_tr) {
  long t = (long)blockIdx.x * 8 + (threadIdx.x >> 5);
  int per = Mt2 * Nt4;
  int b = (int)(t / per); int rem = (int)(t % per);
  int tm2 = rem / Nt4, tn4 = rem % Nt4;
  const bf16_t* A  = Aall + b * sA + (long)tm2 * 32 * lda;
  const bf16_t* Bt = Ball + b * sB + (long)tn4 * 64 * ldb;
  v8f acc[2][4] = {};
  wmma_core_2x4(A, lda, Bt, ldb, K, acc);
  int lane = threadIdx.x & 31, hi2 = lane >> 4, mr = lane & 15;
#pragma unroll
  for (int u = 0; u < 2; ++u) {
    int m0 = tm2 * 32 + u * 16 + hi2 * 8;
#pragma unroll
    for (int s = 0; s < 4; ++s) {
      int col = (tn4 * 4 + s) * 16 + mr;
      if (RM) {
        bf16_t* C = RM + b * sRM;
#pragma unroll
        for (int r = 0; r < 8; ++r) {
          int row = m0 + r;
          C[(long)row * ldrm + col] = (bf16_t)(s_rm * acc[u][s][r] + ((row == col) ? d_rm : 0.f));
        }
      }
      if (TR) {
        v8bf pk;
#pragma unroll
        for (int r = 0; r < 8; ++r) {
          int row = m0 + r;
          pk[r] = (bf16_t)(s_tr * acc[u][s][r] + ((row == col) ? d_tr : 0.f));
        }
        *(v8bf*)(TR + b * sTR + (long)col * ldtr + m0) = pk;
      }
    }
  }
}

// ---------------- softmax over rows of 256 (wave per row)
__global__ void k_softmax256(const float* __restrict__ S, bf16_t* __restrict__ Ob,
                             float* __restrict__ Of) {
  long row = (long)blockIdx.x * 8 + (threadIdx.x >> 5);
  int lane = threadIdx.x & 31;
  const float* p = S + row * 256 + lane * 8;
  float x[8];
#pragma unroll
  for (int i = 0; i < 8; ++i) x[i] = p[i];
  float m = x[0];
#pragma unroll
  for (int i = 1; i < 8; ++i) m = fmaxf(m, x[i]);
  m = wave_max(m);
  float s = 0.f;
#pragma unroll
  for (int i = 0; i < 8; ++i) { x[i] = __expf(x[i] - m); s += x[i]; }
  s = wave_sum(s);
  float inv = 1.f / s;
  if (Of) {
    float* q = Of + row * 256 + lane * 8;
#pragma unroll
    for (int i = 0; i < 8; ++i) q[i] = x[i] * inv;
  }
  if (Ob) {
    v8bf pk;
#pragma unroll
    for (int i = 0; i < 8; ++i) pk[i] = (bf16_t)(x[i] * inv);
    *(v8bf*)(Ob + row * 256 + lane * 8) = pk;
  }
}

// ---------------- softmax over rows of 4096 (block per row)
__global__ void k_softmax4096(const float* __restrict__ S, bf16_t* __restrict__ O) {
  __shared__ float red[8];
  long row = blockIdx.x;
  int tid = threadIdx.x, lane = tid & 31, w = tid >> 5;
  const float* p = S + row * 4096 + tid * 16;
  float x[16];
#pragma unroll
  for (int i = 0; i < 16; ++i) x[i] = p[i];
  float m = x[0];
#pragma unroll
  for (int i = 1; i < 16; ++i) m = fmaxf(m, x[i]);
  m = wave_max(m);
  if (lane == 0) red[w] = m;
  __syncthreads();
  float mm = red[0];
#pragma unroll
  for (int i = 1; i < 8; ++i) mm = fmaxf(mm, red[i]);
  __syncthreads();
  float s = 0.f;
#pragma unroll
  for (int i = 0; i < 16; ++i) { x[i] = __expf(x[i] - mm); s += x[i]; }
  s = wave_sum(s);
  if (lane == 0) red[w] = s;
  __syncthreads();
  float ss = 0.f;
#pragma unroll
  for (int i = 0; i < 8; ++i) ss += red[i];
  float inv = 1.f / ss;
  bf16_t* q = O + row * 4096 + tid * 16;
  v8bf pk0, pk1;
#pragma unroll
  for (int i = 0; i < 8; ++i) { pk0[i] = (bf16_t)(x[i] * inv); pk1[i] = (bf16_t)(x[i + 8] * inv); }
  *(v8bf*)q = pk0; *(v8bf*)(q + 8) = pk1;
}

// ---------------- pinv seed: r = max over all column sums (row sums == 1 after softmax)
__global__ void k_init_r(float* rs) { *rs = 0.f; }
__global__ void k_colmax(const float* __restrict__ a2f, float* __restrict__ rs) {
  int idx = blockIdx.x * 256 + threadIdx.x;                      // 32*256
  int bh = idx >> 8, j = idx & 255;
  const float* p = a2f + (long)bh * 65536 + j;
  float s = 0.f;
  for (int i = 0; i < 256; ++i) s += p[(long)i * 256];
  atomicMax((int*)rs, __float_as_int(s));                        // all values > 0
}
__global__ void k_z0(const float* __restrict__ a2f, const float* __restrict__ rs,
                     bf16_t* __restrict__ zr, bf16_t* __restrict__ zt) {
  long idx = (long)blockIdx.x * 256 + threadIdx.x;               // 32*256*256, [bh][i][j]
  int j = idx & 255; int i = (int)((idx >> 8) & 255); int bh = (int)(idx >> 16);
  float inv = 1.f / *rs;
  zr[idx] = (bf16_t)(a2f[(long)bh * 65536 + (long)j * 256 + i] * inv);  // z = x^T / r
  zt[idx] = (bf16_t)(a2f[(long)bh * 65536 + (long)i * 256 + j] * inv);  // z^T
}

// ---------------- out-heads GEMM: attn1[bh][4096][256] x w1T[bh][64][256] -> OUTH[b][n][h*64+d]
__global__ __launch_bounds__(256, 1)
void k_gemm_outh(const bf16_t* __restrict__ attn1, const bf16_t* __restrict__ w1T,
                 float* __restrict__ outh) {
  long t = (long)blockIdx.x * 8 + (threadIdx.x >> 5);            // 32*128 wide tiles
  int bh = (int)(t / 128); int tm2 = (int)(t % 128);
  const bf16_t* A  = attn1 + (long)bh * 4096 * 256 + (long)tm2 * 32 * 256;
  const bf16_t* Bt = w1T   + (long)bh * 64 * 256;
  v8f acc[2][4] = {};
  wmma_core_2x4(A, 256, Bt, 256, 256, acc);
  int lane = threadIdx.x & 31, hi2 = lane >> 4, mr = lane & 15;
  int bi = bh >> 3, head = bh & 7;
#pragma unroll
  for (int u = 0; u < 2; ++u) {
#pragma unroll
    for (int s = 0; s < 4; ++s) {
      int dc = s * 16 + mr;
#pragma unroll
      for (int r = 0; r < 8; ++r) {
        int ni = tm2 * 32 + u * 16 + hi2 * 8 + r;
        outh[((long)(bi * 4096 + ni)) * 512 + head * 64 + dc] = acc[u][s][r];
      }
    }
  }
}

// ---------------- depthwise 33-tap conv residual over n, fused add + bf16 repack
__global__ void k_conv(const bf16_t* __restrict__ vb, const float* __restrict__ wres,
                       const float* __restrict__ outh, bf16_t* __restrict__ a2o) {
  long idx = (long)blockIdx.x * 256 + threadIdx.x;               // 16384*512
  int dc = idx & 63; int head = (int)((idx >> 6) & 7);
  long ig = idx >> 9; int bi = (int)(ig >> 12); int ni = (int)(ig & 4095);
  long vbase = ((long)(bi * 8 + head)) * 4096 * 64 + dc;
  float s = 0.f;
#pragma unroll
  for (int tt = 0; tt < 33; ++tt) {
    int nn = ni + tt - 16;
    if (nn >= 0 && nn < 4096) s += (float)vb[vbase + (long)nn * 64] * wres[head * 33 + tt];
  }
  a2o[idx] = (bf16_t)(outh[idx] + s);
}

// ---------------- final projection: [16384x512] x [512x512] + bias -> d_out (f32)
__global__ __launch_bounds__(256, 1)
void k_gemm_final(const bf16_t* __restrict__ a2o, const bf16_t* __restrict__ woT,
                  const float* __restrict__ bout, float* __restrict__ out) {
  long t = (long)blockIdx.x * 8 + (threadIdx.x >> 5);            // 512*8 wide tiles
  int tn4 = (int)(t % 8), tm2 = (int)(t / 8);
  const bf16_t* A  = a2o + (long)tm2 * 32 * 512;
  const bf16_t* Bt = woT + (long)tn4 * 64 * 512;
  v8f acc[2][4] = {};
  wmma_core_2x4(A, 512, Bt, 512, 512, acc);
  int lane = threadIdx.x & 31, hi2 = lane >> 4, mr = lane & 15;
#pragma unroll
  for (int u = 0; u < 2; ++u) {
#pragma unroll
    for (int s = 0; s < 4; ++s) {
      int j = (tn4 * 4 + s) * 16 + mr;
      float bj = bout[j];
#pragma unroll
      for (int r = 0; r < 8; ++r)
        out[(long)(tm2 * 32 + u * 16 + hi2 * 8 + r) * 512 + j] = acc[u][s][r] + bj;
    }
  }
}

extern "C" void kernel_launch(void* const* d_in, const int* in_sizes, int n_in,
                              void* d_out, int out_size, void* d_ws, size_t ws_size,
                              hipStream_t stream) {
  (void)in_sizes; (void)n_in; (void)out_size; (void)ws_size;
  const float* x     = (const float*)d_in[0];
  const float* w_qkv = (const float*)d_in[1];
  const float* w_out = (const float*)d_in[2];
  const float* b_out = (const float*)d_in[3];
  const float* w_res = (const float*)d_in[4];
  float* out = (float*)d_out;

  char* w = (char*)d_ws;
  size_t off = 0;
  auto take = [&](size_t bytes) -> char* {
    char* p = w + off; off = (off + bytes + 255) & ~(size_t)255; return p;
  };
  bf16_t* XB  = (bf16_t*)take((size_t)NT_ * DIM_ * 2);
  bf16_t* WQT = (bf16_t*)take((size_t)1536 * 512 * 2);
  bf16_t* QB  = (bf16_t*)take((size_t)BH_ * N_ * D_ * 2);
  bf16_t* KB  = (bf16_t*)take((size_t)BH_ * N_ * D_ * 2);
  bf16_t* VB  = (bf16_t*)take((size_t)BH_ * N_ * D_ * 2);
  bf16_t* VT  = (bf16_t*)take((size_t)BH_ * D_ * N_ * 2);
  bf16_t* QL  = (bf16_t*)take((size_t)BH_ * M_ * D_ * 2);
  bf16_t* KL  = (bf16_t*)take((size_t)BH_ * M_ * D_ * 2);
  float*  SIM = (float*) take((size_t)BH_ * N_ * M_ * 4);   // reused for sim1/sim2/sim3
  bf16_t* AT1 = (bf16_t*)take((size_t)BH_ * N_ * M_ * 2);
  bf16_t* AT3 = (bf16_t*)take((size_t)BH_ * M_ * N_ * 2);
  float*  A2F = (float*) take((size_t)BH_ * M_ * M_ * 4);
  bf16_t* A2B = (bf16_t*)take((size_t)BH_ * M_ * M_ * 2);
  bf16_t* Z1R = (bf16_t*)take((size_t)BH_ * M_ * M_ * 2);
  bf16_t* Z1T = (bf16_t*)take((size_t)BH_ * M_ * M_ * 2);
  bf16_t* Z2R = (bf16_t*)take((size_t)BH_ * M_ * M_ * 2);
  bf16_t* Z2T = (bf16_t*)take((size_t)BH_ * M_ * M_ * 2);
  bf16_t* XZ  = (bf16_t*)take((size_t)BH_ * M_ * M_ * 2);
  bf16_t* TA  = (bf16_t*)take((size_t)BH_ * M_ * M_ * 2);
  bf16_t* TB  = (bf16_t*)take((size_t)BH_ * M_ * M_ * 2);
  bf16_t* A3VT= (bf16_t*)take((size_t)BH_ * D_ * M_ * 2);
  bf16_t* W1T = (bf16_t*)take((size_t)BH_ * D_ * M_ * 2);
  float*  OUTH= (float*) take((size_t)NT_ * INNER_ * 4);
  bf16_t* A2O = (bf16_t*)take((size_t)NT_ * INNER_ * 2);
  bf16_t* WOT = (bf16_t*)take((size_t)512 * 512 * 2);
  float*  RS  = (float*) take(256);

  // prep
  k_cvt_x   <<<32768, 256, 0, stream>>>(x, XB);
  k_tr_w    <<<3072,  256, 0, stream>>>(w_qkv, WQT);
  k_tr_wout <<<1024,  256, 0, stream>>>(w_out, WOT);
  // qkv projection: 512*24 wide tiles / 8 waves
  k_gemm_qkv<<<1536, 256, 0, stream>>>(XB, WQT, QB, KB, VB, VT);
  k_landmarks<<<2048, 256, 0, stream>>>(QB, KB, QL, KL);
  // sim1 = q @ k_land^T  -> softmax -> attn1 (bf16)   (Mt2=128, Nt4=4)
  k_gemm_f32<<<2048, 256, 0, stream>>>(QB, KL, SIM, 128, 4, 64, 64, 64, 256,
                                       (long)N_ * D_, (long)M_ * D_, (long)N_ * M_);
  k_softmax256<<<16384, 256, 0, stream>>>(SIM, AT1, nullptr);
  // sim2 = q_land @ k_land^T -> softmax -> attn2 (f32 + bf16)   (Mt2=8, Nt4=4)
  k_gemm_f32<<<128, 256, 0, stream>>>(QL, KL, SIM, 8, 4, 64, 64, 64, 256,
                                      (long)M_ * D_, (long)M_ * D_, (long)M_ * M_);
  k_softmax256<<<1024, 256, 0, stream>>>(SIM, A2B, A2F);
  // pinv seed: z0 = x^T / max_colsum
  k_init_r<<<1, 1, 0, stream>>>(RS);
  k_colmax<<<32, 256, 0, stream>>>(A2F, RS);
  k_z0<<<8192, 256, 0, stream>>>(A2F, RS, Z1R, Z1T);
  // sim3 = q_land @ k^T -> softmax -> attn3 (bf16)    (Mt2=8, Nt4=64)
  k_gemm_f32<<<2048, 256, 0, stream>>>(QL, KB, SIM, 8, 64, 64, 64, 64, 4096,
                                       (long)M_ * D_, (long)N_ * D_, (long)M_ * N_);
  k_softmax4096<<<8192, 256, 0, stream>>>(SIM, AT3);
  // Moore-Penrose iterations: z = 0.25 z (13I - xz(15I - xz(7I - xz)))   (Mt2=8, Nt4=4)
  bf16_t *zr = Z1R, *zt = Z1T, *z2r = Z2R, *z2t = Z2T;
  const long SQ = (long)M_ * M_;
  for (int it = 0; it < 6; ++it) {
    k_gemm_b16<<<128, 256, 0, stream>>>(A2B, zt, XZ, TA, 8, 4, 256, 256, 256,
                                        SQ, SQ, SQ, 256, SQ, 256, 1.f, 0.f, -1.f, 7.f);
    k_gemm_b16<<<128, 256, 0, stream>>>(XZ, TA, nullptr, TB, 8, 4, 256, 256, 256,
                                        SQ, SQ, 0, 256, SQ, 256, 0.f, 0.f, -1.f, 15.f);
    k_gemm_b16<<<128, 256, 0, stream>>>(XZ, TB, nullptr, TA, 8, 4, 256, 256, 256,
                                        SQ, SQ, 0, 256, SQ, 256, 0.f, 0.f, -1.f, 13.f);
    k_gemm_b16<<<128, 256, 0, stream>>>(zr, TA, z2r, z2t, 8, 4, 256, 256, 256,
                                        SQ, SQ, SQ, 256, SQ, 256, 0.25f, 0.f, 0.25f, 0.f);
    bf16_t* t0 = zr; zr = z2r; z2r = t0;
    bf16_t* t1 = zt; zt = z2t; z2t = t1;
  }
  // attn3 @ v  -> (store transposed)   (Mt2=8, Nt4=1)
  k_gemm_b16<<<32, 256, 0, stream>>>(AT3, VT, nullptr, A3VT, 8, 1, 4096, 4096, 4096,
                                     (long)M_ * N_, (long)D_ * N_, 0, 256,
                                     (long)D_ * M_, 256, 0.f, 0.f, 1.f, 0.f);
  // z @ (attn3 v) -> (store transposed)   (Mt2=8, Nt4=1)
  k_gemm_b16<<<32, 256, 0, stream>>>(zr, A3VT, nullptr, W1T, 8, 1, 256, 256, 256,
                                     SQ, (long)D_ * M_, 0, 256,
                                     (long)D_ * M_, 256, 0.f, 0.f, 1.f, 0.f);
  // attn1 @ w1 -> OUTH [b][n][h*64+d]   (Mt2=128, Nt4=1)
  k_gemm_outh<<<512, 256, 0, stream>>>(AT1, W1T, OUTH);
  // + depthwise conv residual, repack to bf16
  k_conv<<<32768, 256, 0, stream>>>(VB, w_res, OUTH, A2O);
  // final projection with bias   (Mt2=512, Nt4=8)
  k_gemm_final<<<512, 256, 0, stream>>>(A2O, WOT, b_out, out);
}
